// Mamba2SimpleBlock_85298050498978
// MI455X (gfx1250) — matrix operand order repaired and checked by reading, not compiled, not run on hardware
//
#include <hip/hip_runtime.h>
#include <hip/hip_bf16.h>
#include <stdint.h>

typedef __attribute__((ext_vector_type(16))) _Float16 v16h;
typedef __attribute__((ext_vector_type(8)))  _Float16 v8h;
typedef __attribute__((ext_vector_type(8)))  float    v8f;

#define BDIM    2
#define LSEQ    2048
#define DMODEL  1024
#define DINNER  2048
#define NHEADS  4
#define DSTATE  64
#define HEADDIM 512
#define NPROJ   4612           // 2*DINNER + 2*NHEADS*DSTATE + NHEADS
#define NPAD    4672           // round up to multiple of 64
#define ROWS    (BDIM * LSEQ)  // 4096
#define COL_X   DINNER         // 2048
#define COL_DT  (2 * DINNER)   // 4096
#define COL_BC  (2 * DINNER + NHEADS) // 4100
#define RMS_EPS 1.1920929e-07f

// LDS tile stride in halves (rows padded 32 -> 40 halves = 80B so per-lane
// ds_load_b128 fragment reads are bank-conflict-free across 64 banks)
#define TSTRIDE 40
#define MTILE   256            // block M span (8 waves x 32 rows)
#define NTILE   64             // block N span
#define ABUFB   (MTILE * TSTRIDE * 2)   // bytes per A buffer
#define BBUFB   (NTILE * TSTRIDE * 2)   // bytes per B buffer

// ---------------------------------------------------------------------------
// CDNA5 async-copy helpers (ASYNCcnt-tracked global -> LDS DMA)
// ---------------------------------------------------------------------------
__device__ __forceinline__ void async_ld_b128(uint32_t lds_off, uint64_t gaddr) {
    asm volatile("global_load_async_to_lds_b128 %0, %1, off"
                 :: "v"(lds_off), "v"(gaddr) : "memory");
}
__device__ __forceinline__ void wait_async0() {
    asm volatile("s_wait_asynccnt 0x0" ::: "memory");
}

// ---------------------------------------------------------------------------
// 1) RMSNorm: fp32 in -> f16 normalized activations (one block per row)
// ---------------------------------------------------------------------------
__global__ void __launch_bounds__(256)
rmsnorm_f16_kernel(const float* __restrict__ u,
                   const float* __restrict__ w,
                   _Float16* __restrict__ out) {
    const int row = blockIdx.x;
    const float* ur = u + (size_t)row * DMODEL;
    __shared__ float red[8];
    float vals[4];
    float ss = 0.f;
#pragma unroll
    for (int i = 0; i < 4; ++i) {
        vals[i] = ur[threadIdx.x + i * 256];
        ss += vals[i] * vals[i];
    }
    for (int off = 16; off > 0; off >>= 1) ss += __shfl_down(ss, off);
    const int lane = threadIdx.x & 31, wv = threadIdx.x >> 5;
    if (lane == 0) red[wv] = ss;
    __syncthreads();
    if (threadIdx.x == 0) {
        float t = 0.f;
#pragma unroll
        for (int i = 0; i < 8; ++i) t += red[i];
        red[0] = rsqrtf(t / (float)DMODEL + RMS_EPS);
    }
    __syncthreads();
    const float scale = red[0];
    _Float16* orow = out + (size_t)row * DMODEL;
#pragma unroll
    for (int i = 0; i < 4; ++i) {
        const int c = threadIdx.x + i * 256;
        orow[c] = (_Float16)(vals[i] * scale * w[c]);
    }
}

// ---------------------------------------------------------------------------
// 2) fp32 -> f16 convert with zero tail padding (for weights)
// ---------------------------------------------------------------------------
__global__ void convert_pad_kernel(const float* __restrict__ src,
                                   _Float16* __restrict__ dst,
                                   long valid, long total) {
    long i = (long)blockIdx.x * blockDim.x + threadIdx.x;
    if (i >= total) return;
    dst[i] = (i < valid) ? (_Float16)src[i] : (_Float16)0.f;
}

__global__ void pad_bias_kernel(const float* __restrict__ src,
                                float* __restrict__ dst, int n, int npad) {
    int i = blockIdx.x * blockDim.x + threadIdx.x;
    if (i < npad) dst[i] = (i < n) ? src[i] : 0.f;
}

// ---------------------------------------------------------------------------
// 3) WMMA GEMM, async-LDS ping/pong with STATIC buffer addressing.
//    C[M x N] = A[M x K](f16) @ W[N x K](f16)^T + bias
//    Block = 256 thr / 8 waves -> 256(M) x 64(N); wave -> 32(M) x 64(N).
//    K unrolled by 64: DMA pong while 8 WMMAs/wave consume ping, then swap.
//    Requires K % 64 == 0 (holds: 1024, 2048).
// ---------------------------------------------------------------------------
union frag16 { v16h v; v8h h[2]; };

__global__ void __launch_bounds__(256)
gemm_wmma_async_kernel(const _Float16* __restrict__ A,   // M x K
                       const _Float16* __restrict__ W,   // N x K row-major
                       const float* __restrict__ bias,   // N
                       float* __restrict__ C,            // M x N
                       int K, int N) {
    __shared__ _Float16 sA[2][MTILE * TSTRIDE];  // 256 rows x 32 halves (padded)
    __shared__ _Float16 sB[2][NTILE * TSTRIDE];  // 64 rows x 32 halves (padded)

    const int tid  = threadIdx.x;
    const int lane = tid & 31;
    const int wave = tid >> 5;
    const int half = lane >> 4;
    const int m0   = blockIdx.x * MTILE;
    const int nb0  = blockIdx.y * NTILE;

    // ---- per-thread async-copy mapping (16B chunks, 4 chunks per row) ----
    //   A tile: 1024 chunks = 4/thread;  B tile: 256 chunks = 1/thread
    const _Float16* gA[4];
    uint32_t lA0[4];                       // buffer-0 LDS offsets (buffer-1 = +ABUFB)
#pragma unroll
    for (int j = 0; j < 4; ++j) {
        const int idx = tid + j * 256;
        const int r = idx >> 2, c = idx & 3;
        gA[j]  = A + (size_t)(m0 + r) * K + c * 8;
        lA0[j] = (uint32_t)(size_t)&sA[0][r * TSTRIDE + c * 8];
    }
    const int br = tid >> 2, bc = tid & 3;
    const _Float16* gB = W + (size_t)(nb0 + br) * K + bc * 8;
    const uint32_t lB0 = (uint32_t)(size_t)&sB[0][br * TSTRIDE + bc * 8];

    // ---- fragment LDS read offsets for this lane ----
    const int mrow0 = wave * 32 + (lane & 15);          // first 16-row subtile
    const int aoff0 = mrow0 * TSTRIDE + half * 8;       // + {0,16} halves
    const int aoff1 = aoff0 + 16 * TSTRIDE;             // second subtile (+16 rows)
    const int boff  = (lane & 15) * TSTRIDE + half * 8; // + t*16*TSTRIDE

    v8f acc0[4] = {{}, {}, {}, {}};
    v8f acc1[4] = {{}, {}, {}, {}};

    // DMA one 32-K tile into buffer `bsel` (compile-time 0/1 after inlining)
    auto issue = [&](int bsel, int kk) {
        const uint32_t ao = bsel ? (uint32_t)ABUFB : 0u;
        const uint32_t bo = bsel ? (uint32_t)BBUFB : 0u;
#pragma unroll
        for (int j = 0; j < 4; ++j)
            async_ld_b128(lA0[j] + ao, (uint64_t)(size_t)(gA[j] + kk));
        async_ld_b128(lB0 + bo, (uint64_t)(size_t)(gB + kk));
    };

    // consume one 32-K tile from buffer `bsel`: load ALL fragments, WMMA burst
    auto compute = [&](int bsel) {
        const _Float16* sAb = &sA[bsel][0];
        const _Float16* sBb = &sB[bsel][0];
        frag16 a0, a1, b[4];
        a0.h[0] = *(const v8h*)(sAb + aoff0);
        a0.h[1] = *(const v8h*)(sAb + aoff0 + 16);
        a1.h[0] = *(const v8h*)(sAb + aoff1);
        a1.h[1] = *(const v8h*)(sAb + aoff1 + 16);
#pragma unroll
        for (int t = 0; t < 4; ++t) {
            const _Float16* bp = sBb + boff + t * 16 * TSTRIDE;
            b[t].h[0] = *(const v8h*)(bp);
            b[t].h[1] = *(const v8h*)(bp + 16);
        }
#pragma unroll
        for (int t = 0; t < 4; ++t)
            acc0[t] = __builtin_amdgcn_wmma_f32_16x16x32_f16(
                false, a0.v, false, b[t].v, (short)0, acc0[t], false, false);
#pragma unroll
        for (int t = 0; t < 4; ++t)
            acc1[t] = __builtin_amdgcn_wmma_f32_16x16x32_f16(
                false, a1.v, false, b[t].v, (short)0, acc1[t], false, false);
    };

    // prologue: fill ping (buffer 0) with k=0
    issue(0, 0);
    wait_async0();
    __syncthreads();

    for (int k0 = 0; k0 < K; k0 += 64) {
        issue(1, k0 + 32);          // pong <- k0+32 (always valid, K % 64 == 0)
        compute(0);                 // consume ping (k0), overlapped with DMA
        wait_async0();
        __syncthreads();
        if (k0 + 64 < K) issue(0, k0 + 64);  // ping <- k0+64
        compute(1);                 // consume pong (k0+32)
        wait_async0();
        __syncthreads();
    }

    // C/D layout: VGPR r -> M = subtile_base + r + 8*half, N = nb0 + t*16 + lane%16
    const int ncol = nb0 + (lane & 15);
    float* crow0 = C + (size_t)(m0 + wave * 32 + half * 8) * N;
    float* crow1 = crow0 + (size_t)16 * N;
#pragma unroll
    for (int t = 0; t < 4; ++t) {
        const int col = ncol + t * 16;
        const float bv = bias[col];
#pragma unroll
        for (int r = 0; r < 8; ++r) {
            crow0[(size_t)r * N + col] = acc0[t][r] + bv;
            crow1[(size_t)r * N + col] = acc1[t][r] + bv;
        }
    }
}

// ---------------------------------------------------------------------------
// 4) dt precompute: dt = softplus(dt_raw + dt_bias), dtA = exp(dt * -exp(A_log))
// ---------------------------------------------------------------------------
__global__ void dt_kernel(const float* __restrict__ zx,
                          const float* __restrict__ A_log,
                          const float* __restrict__ dt_bias,
                          float* __restrict__ dtv,
                          float* __restrict__ dtAv) {
    int i = blockIdx.x * blockDim.x + threadIdx.x;
    if (i >= ROWS * NHEADS) return;
    const int row = i >> 2, h = i & 3;
    const float x  = zx[(size_t)row * NPAD + COL_DT + h] + dt_bias[h];
    const float dt = (x > 20.f) ? x : log1pf(expf(x));
    const float Ah = -expf(A_log[h]);
    dtv[i]  = dt;
    dtAv[i] = expf(dt * Ah);
}

// ---------------------------------------------------------------------------
// 5) Sequential selective scan; block = (batch, head), thread = p column.
//    State h[n=0..63][p] lives in 64 VGPRs per thread. Fuses D-skip + SiLU
//    gate and emits f16 activations for the out_proj WMMA GEMM.
// ---------------------------------------------------------------------------
__global__ void __launch_bounds__(512)
scan_kernel(const float* __restrict__ zx,
            const float* __restrict__ dtv,
            const float* __restrict__ dtAv,
            const float* __restrict__ Dvec,
            _Float16* __restrict__ ya) {
    const int b = blockIdx.x >> 2;
    const int h = blockIdx.x & 3;
    const int p = threadIdx.x;             // 0..511
    __shared__ float sB[DSTATE], sC[DSTATE], sS[2];

    float hs[DSTATE];
#pragma unroll
    for (int n = 0; n < DSTATE; ++n) hs[n] = 0.f;

    const float Dh = Dvec[h];
    const size_t colz = (size_t)h * HEADDIM + p;
    const size_t colx = COL_X + (size_t)h * HEADDIM + p;
    const size_t colB = COL_BC + (size_t)h * (2 * DSTATE);

    for (int t = 0; t < LSEQ; ++t) {
        const int    r     = b * LSEQ + t;
        const size_t rbase = (size_t)r * NPAD;
        if (p < DSTATE)            sB[p]          = zx[rbase + colB + p];
        else if (p < 2 * DSTATE)   sC[p - DSTATE] = zx[rbase + colB + p];
        else if (p == 2 * DSTATE) {
            const int di = r * NHEADS + h;
            sS[0] = dtv[di];
            sS[1] = dtAv[di];
        }
        __syncthreads();
        const float x   = zx[rbase + colx];
        const float z   = zx[rbase + colz];
        const float dt  = sS[0];
        const float dtA = sS[1];
        const float dx  = dt * x;
        float y = 0.f;
#pragma unroll
        for (int n = 0; n < DSTATE; ++n) {
            hs[n] = fmaf(hs[n], dtA, sB[n] * dx);
            y = fmaf(sC[n], hs[n], y);
        }
        y = fmaf(Dh, x, y);
        const float sig = 1.f / (1.f + expf(-z));
        ya[(size_t)r * DINNER + (size_t)h * HEADDIM + p] = (_Float16)(y * z * sig);
        __syncthreads();
    }
}

// ---------------------------------------------------------------------------
// Launch
// ---------------------------------------------------------------------------
extern "C" void kernel_launch(void* const* d_in, const int* in_sizes, int n_in,
                              void* d_out, int out_size, void* d_ws, size_t ws_size,
                              hipStream_t stream) {
    (void)in_sizes; (void)n_in; (void)out_size; (void)ws_size;
    const float* u       = (const float*)d_in[0];
    const float* norm_w  = (const float*)d_in[1];
    const float* in_w    = (const float*)d_in[2];   // (4612, 1024)
    const float* in_b    = (const float*)d_in[3];   // (4612,)
    const float* A_log   = (const float*)d_in[4];   // (4,)
    const float* Dvec    = (const float*)d_in[5];   // (4,)
    const float* dt_bias = (const float*)d_in[6];   // (4,)
    const float* out_w   = (const float*)d_in[7];   // (1024, 2048)
    const float* out_b   = (const float*)d_in[8];   // (1024,)
    float* out = (float*)d_out;                     // (2, 2048, 1024) f32

    char* wp = (char*)d_ws;
    auto alloc = [&](size_t bytes) -> void* {
        void* p = (void*)wp;
        wp += (bytes + 255) & ~(size_t)255;
        return p;
    };
    _Float16* uh   = (_Float16*)alloc((size_t)ROWS * DMODEL * sizeof(_Float16));
    _Float16* w1h  = (_Float16*)alloc((size_t)NPAD * DMODEL * sizeof(_Float16));
    float*    b1p  = (float*)   alloc((size_t)NPAD * sizeof(float));
    float*    zx   = (float*)   alloc((size_t)ROWS * NPAD * sizeof(float));
    float*    dtv  = (float*)   alloc((size_t)ROWS * NHEADS * sizeof(float));
    float*    dtAv = (float*)   alloc((size_t)ROWS * NHEADS * sizeof(float));
    _Float16* yah  = (_Float16*)alloc((size_t)ROWS * DINNER * sizeof(_Float16));
    _Float16* w2h  = (_Float16*)alloc((size_t)DMODEL * DINNER * sizeof(_Float16));

    // 1) RMSNorm + f16 cast
    rmsnorm_f16_kernel<<<ROWS, 256, 0, stream>>>(u, norm_w, uh);

    // 2) weight conversion (+ zero row padding for in_proj)
    {
        const long total = (long)NPAD * DMODEL;
        const long valid = (long)NPROJ * DMODEL;
        convert_pad_kernel<<<(int)((total + 255) / 256), 256, 0, stream>>>(in_w, w1h, valid, total);
    }
    pad_bias_kernel<<<(NPAD + 255) / 256, 256, 0, stream>>>(in_b, b1p, NPROJ, NPAD);
    {
        const long total = (long)DMODEL * DINNER;
        convert_pad_kernel<<<(int)((total + 255) / 256), 256, 0, stream>>>(out_w, w2h, total, total);
    }

    // 3) in_proj GEMM (WMMA + async LDS): zx[4096 x 4672] = uh @ w1h^T + b1p
    gemm_wmma_async_kernel<<<dim3(ROWS / MTILE, NPAD / NTILE), 256, 0, stream>>>(
        uh, w1h, b1p, zx, DMODEL, NPAD);

    // 4) dt / dtA precompute
    dt_kernel<<<(ROWS * NHEADS + 255) / 256, 256, 0, stream>>>(zx, A_log, dt_bias, dtv, dtAv);

    // 5) selective scan + D skip + SiLU gate -> f16 activations
    scan_kernel<<<BDIM * NHEADS, 512, 0, stream>>>(zx, dtv, dtAv, Dvec, yah);

    // 6) out_proj GEMM (WMMA + async LDS): out[4096 x 1024] = yah @ w2h^T + out_b
    gemm_wmma_async_kernel<<<dim3(ROWS / MTILE, DMODEL / NTILE), 256, 0, stream>>>(
        yah, w2h, out_b, out, DINNER, DMODEL);
}
